// FCM_64836826300505
// MI455X (gfx1250) — matrix operand order repaired and checked by reading, not compiled, not run on hardware
//
#include <hip/hip_runtime.h>
#include <hip/hip_bf16.h>

// Problem constants
#define NROW    8192
#define KNN     16
#define DIM     256
#define LIST    17        // K+1 (self included, dropped at the end)
#define CHUNK   32        // columns staged in LDS per pipeline stage (= 2 WMMA tiles)
#define CSTRIDE 272       // padded ushort elements per staged column (bank-conflict pad)

typedef __bf16        v16bf __attribute__((ext_vector_type(16)));
typedef float         v8f   __attribute__((ext_vector_type(8)));
typedef unsigned int  u32x4 __attribute__((ext_vector_type(4)));

union FragU { v16bf bf; u32x4 q[2]; };

#define INFF __builtin_huge_valf()

// ---------------------------------------------------------------------------
// Async global->LDS copy (CDNA5 GLOBAL_LOAD_ASYNC_TO_LDS_B128, ASYNCcnt).
// vdst = LDS byte offset VGPR, vaddr = 64-bit global address, off = no saddr.
// ---------------------------------------------------------------------------
#define ASYNC_LD_B128(ldsoff, gaddr, imm)                                     \
  asm volatile("global_load_async_to_lds_b128 %0, %1, off offset:" #imm      \
               :: "v"(ldsoff), "v"(gaddr) : "memory")

__device__ __forceinline__ void wait_async0() {
#if __has_builtin(__builtin_amdgcn_s_wait_asynccnt)
  __builtin_amdgcn_s_wait_asynccnt(0);
#else
  asm volatile("s_wait_asynccnt 0" ::: "memory");
#endif
}

// ---------------------------------------------------------------------------
// Kernel A: f32 -> bf16 (RNE) copy + row squared norms
// ---------------------------------------------------------------------------
__global__ __launch_bounds__(256) void prep_kernel(const float* __restrict__ feats,
                                                   unsigned short* __restrict__ fbf,
                                                   float* __restrict__ sq) {
  const int row = blockIdx.x;
  const int t   = threadIdx.x;
  float x = feats[(size_t)row * DIM + t];
  unsigned int u = __float_as_uint(x);
  unsigned int r = (u + 0x7FFFu + ((u >> 16) & 1u)) >> 16;   // round-to-nearest-even bf16
  fbf[(size_t)row * DIM + t] = (unsigned short)r;
  float ss = x * x;
  #pragma unroll
  for (int off = 16; off > 0; off >>= 1) ss += __shfl_down(ss, off, 32);
  __shared__ float red[8];
  if ((t & 31) == 0) red[t >> 5] = ss;
  __syncthreads();
  if (t == 0) {
    float tot = 0.0f;
    #pragma unroll
    for (int wv = 0; wv < 8; ++wv) tot += red[wv];
    sq[row] = tot;
  }
}

// ---------------------------------------------------------------------------
// Sorted (ascending) insertion into a 17-entry register list. Fully unrolled
// static indexing so ld/li stay in VGPRs.
// ---------------------------------------------------------------------------
__device__ __forceinline__ void topk_insert(float* ld, int* li, float d, int idx) {
  float cd = d; int ci = idx;
  #pragma unroll
  for (int p = 0; p < LIST; ++p) {
    bool sw = cd < ld[p];
    float od = ld[p]; int oi = li[p];
    ld[p] = sw ? cd : od;
    li[p] = sw ? ci : oi;
    cd    = sw ? od : cd;
    ci    = sw ? oi : ci;
  }
}

// ---------------------------------------------------------------------------
// Kernel B: fused bf16-WMMA Gram tiles + running per-row top-17 of d^2.
// Block = 128 threads = 4 waves, each wave owns 16 rows (A tile in VGPRs).
// Column chunks are async-staged into double-buffered LDS and shared by all
// 4 waves; loads for chunk i+1 overlap compute on chunk i. Each chunk is two
// 16-col tiles computed as two INDEPENDENT interleaved WMMA chains, with the
// B fragments software-pipelined one k-step ahead so every ds_load has two
// WMMAs of latency cover before its s_wait_dscnt.
// ---------------------------------------------------------------------------
__global__ __launch_bounds__(128) void knn_kernel(const unsigned short* __restrict__ fbf,
                                                  const float* __restrict__ sq,
                                                  int*   __restrict__ knn_idx,
                                                  float* __restrict__ knn_score) {
  __shared__ unsigned short ldsB[2][CHUNK * CSTRIDE];  // 2 x 17408 B staged columns
  __shared__ float ldsSq[2][CHUNK];
  __shared__ float ldsC[4][2][16 * 17];                // per-wave C-tile transpose scratch
  __shared__ float ldsMD[4][16 * LIST];                // merge: lane-half lists (dist)
  __shared__ int   ldsMI[4][16 * LIST];                // merge: lane-half lists (idx)

  const int tid  = threadIdx.x;
  const int wv   = tid >> 5;
  const int lane = tid & 31;
  const int lrow = lane & 15;      // row within 16x16 tile (also B column for B frags)
  const int hf   = lane >> 4;      // lane half
  const int rowBase = blockIdx.x * 64 + wv * 16;
  const int grow = rowBase + lrow;

  // Per-thread staging job: 64 elements (128 B) of one column quarter.
  const int scol = tid >> 2;
  const int sq4  = tid & 3;

  // --- A fragments for this wave's 16 rows: 8 k-steps of 16x32 bf16 ---
  // Layout: lanes 0-15 hold K = k0+0..7 (q0), k0+8..15 (q1);
  // lanes 16-31 hold K = k0+16..23, k0+24..31.
  FragU afrag[8];
  const u32x4* arow = (const u32x4*)(fbf + (size_t)grow * DIM);
  #pragma unroll
  for (int s = 0; s < 8; ++s) {
    afrag[s].q[0] = arow[4 * s + 2 * hf];
    afrag[s].q[1] = arow[4 * s + 2 * hf + 1];
  }
  const float sqr = sq[grow];

  float ld[LIST]; int li[LIST];
  #pragma unroll
  for (int p = 0; p < LIST; ++p) { ld[p] = INFF; li[p] = -1; }

  // --- pipeline prologue: async-stage chunk 0 into buffer 0 ---
  {
    unsigned long long ga =
        (unsigned long long)(fbf + (size_t)scol * DIM + sq4 * 64);
    unsigned lo = (unsigned)(unsigned long long)(void*)
        (&ldsB[0][0] + scol * CSTRIDE + sq4 * 64);
    ASYNC_LD_B128(lo, ga, 0);   ASYNC_LD_B128(lo, ga, 16);
    ASYNC_LD_B128(lo, ga, 32);  ASYNC_LD_B128(lo, ga, 48);
    ASYNC_LD_B128(lo, ga, 64);  ASYNC_LD_B128(lo, ga, 80);
    ASYNC_LD_B128(lo, ga, 96);  ASYNC_LD_B128(lo, ga, 112);
    if (tid < CHUNK) ldsSq[0][tid] = sq[tid];
  }
  wait_async0();
  __syncthreads();

  for (int cb = 0; cb < NROW; cb += CHUNK) {
    const int buf = (cb / CHUNK) & 1;

    // --- issue async loads for next chunk into the other buffer ---
    if (cb + CHUNK < NROW) {
      unsigned long long ga =
          (unsigned long long)(fbf + (size_t)(cb + CHUNK + scol) * DIM + sq4 * 64);
      unsigned lo = (unsigned)(unsigned long long)(void*)
          (&ldsB[buf ^ 1][0] + scol * CSTRIDE + sq4 * 64);
      ASYNC_LD_B128(lo, ga, 0);   ASYNC_LD_B128(lo, ga, 16);
      ASYNC_LD_B128(lo, ga, 32);  ASYNC_LD_B128(lo, ga, 48);
      ASYNC_LD_B128(lo, ga, 64);  ASYNC_LD_B128(lo, ga, 80);
      ASYNC_LD_B128(lo, ga, 96);  ASYNC_LD_B128(lo, ga, 112);
      if (tid < CHUNK) ldsSq[buf ^ 1][tid] = sq[cb + CHUNK + tid];
    }

    // --- compute both tiles of the chunk as two independent WMMA chains ---
    // B (32x16 bf16): lanes 0-15 K=k0..k0+15 linear, lanes 16-31 K=k0+16..k0+31
    {
      const u32x4* bp0 = (const u32x4*)(&ldsB[buf][0] + lrow * CSTRIDE);
      const u32x4* bp1 = (const u32x4*)(&ldsB[buf][0] + (16 + lrow) * CSTRIDE);
      v8f c0 = {0.0f, 0.0f, 0.0f, 0.0f, 0.0f, 0.0f, 0.0f, 0.0f};
      v8f c1 = {0.0f, 0.0f, 0.0f, 0.0f, 0.0f, 0.0f, 0.0f, 0.0f};

      FragU b0, b1;
      b0.q[0] = bp0[2 * hf];
      b0.q[1] = bp0[2 * hf + 1];
      b1.q[0] = bp1[2 * hf];
      b1.q[1] = bp1[2 * hf + 1];
      #pragma unroll
      for (int s = 0; s < 8; ++s) {
        FragU nb0, nb1;
        if (s < 7) {                       // prefetch next k-step fragments
          nb0.q[0] = bp0[4 * (s + 1) + 2 * hf];
          nb0.q[1] = bp0[4 * (s + 1) + 2 * hf + 1];
          nb1.q[0] = bp1[4 * (s + 1) + 2 * hf];
          nb1.q[1] = bp1[4 * (s + 1) + 2 * hf + 1];
        }
        c0 = __builtin_amdgcn_wmma_f32_16x16x32_bf16(false, afrag[s].bf,
                                                     false, b0.bf,
                                                     (short)0, c0, false, false);
        c1 = __builtin_amdgcn_wmma_f32_16x16x32_bf16(false, afrag[s].bf,
                                                     false, b1.bf,
                                                     (short)0, c1, false, false);
        if (s < 7) { b0 = nb0; b1 = nb1; } // register rotate (renamed by unroll)
      }

      // --- transpose both C tiles through padded LDS: (M = r + 8*hf, N = lrow) ---
      float* ct0 = &ldsC[wv][0][0];
      float* ct1 = &ldsC[wv][1][0];
      #pragma unroll
      for (int r = 0; r < 8; ++r) {
        ct0[(r + hf * 8) * 17 + lrow] = c0[r];
        ct1[(r + hf * 8) * 17 + lrow] = c1[r];
      }

      // --- candidates: lane processes its own row (lrow), cols hf*8..hf*8+7 ---
      #pragma unroll
      for (int tile = 0; tile < 2; ++tile) {
        const float* ct = tile ? ct1 : ct0;
        const int colBase = cb + tile * 16;
        #pragma unroll
        for (int ci = 0; ci < 8; ++ci) {
          const int n  = hf * 8 + ci;
          const float s  = ct[lrow * 17 + n];
          const float d2 = sqr + ldsSq[buf][tile * 16 + n] - 2.0f * s;
          if (d2 < ld[LIST - 1]) topk_insert(ld, li, d2, colBase + n);
        }
      }
    }

    wait_async0();     // next buffer fully landed (per-wave, in-order ASYNCcnt)
    __syncthreads();   // all waves done reading current buffer + see next one
  }

  // --- merge the two lane-half lists per row (wave-local, DS-ordered) ---
  if (hf == 1) {
    #pragma unroll
    for (int p = 0; p < LIST; ++p) {
      ldsMD[wv][lrow * LIST + p] = ld[p];
      ldsMI[wv][lrow * LIST + p] = li[p];
    }
  }
  if (hf == 0) {
    for (int p = 0; p < LIST; ++p) {
      float d  = ldsMD[wv][lrow * LIST + p];
      int   ix = ldsMI[wv][lrow * LIST + p];
      if (d < ld[LIST - 1]) topk_insert(ld, li, d, ix);
    }
    // drop entry 0 (self), emit K neighbors with their scores
    #pragma unroll
    for (int t = 1; t < LIST; ++t) {
      int   gj = li[t];
      float d  = ld[t];
      float sc = 0.5f * (sqr + sq[gj] - d);   // recover score = feats_i . feats_j
      knn_idx  [(size_t)grow * KNN + (t - 1)] = gj;
      knn_score[(size_t)grow * KNN + (t - 1)] = sc;
    }
  }
}

// ---------------------------------------------------------------------------
// Kernel C: mutual mask, 17-way softmax, aggregation (f32 feats), normalize
// ---------------------------------------------------------------------------
__global__ __launch_bounds__(256) void apply_kernel(const float* __restrict__ feats,
                                                    const int*   __restrict__ knn_idx,
                                                    const float* __restrict__ knn_score,
                                                    float* __restrict__ out) {
  const int i = blockIdx.x;
  const int t = threadIdx.x;
  __shared__ float w[KNN];
  __shared__ int   nj[KNN];
  __shared__ float wdiag;
  __shared__ float red[8];
  __shared__ float scale_s;

  if (t < KNN) {
    int   j = knn_idx [(size_t)i * KNN + t];
    float s = knn_score[(size_t)i * KNN + t];
    bool mutual = false;
    #pragma unroll
    for (int u = 0; u < KNN; ++u) mutual |= (knn_idx[(size_t)j * KNN + u] == i);
    nj[t] = j;
    w[t]  = mutual ? s : -INFF;
  }
  __syncthreads();
  if (t == 0) {
    float m = 1.0f;                         // diagonal logit
    for (int u = 0; u < KNN; ++u) m = fmaxf(m, w[u]);
    float ed = __expf(1.0f - m);
    float denom = ed;
    float e[KNN];
    for (int u = 0; u < KNN; ++u) {
      float x = w[u];
      float v = (x > -INFF) ? __expf(x - m) : 0.0f;
      e[u] = v; denom += v;
    }
    float inv = 1.0f / denom;
    wdiag = ed * inv;
    for (int u = 0; u < KNN; ++u) w[u] = e[u] * inv;
  }
  __syncthreads();
  float acc = feats[(size_t)i * DIM + t] * (1.0f + wdiag);
  #pragma unroll
  for (int u = 0; u < KNN; ++u) {
    float wu = w[u];
    if (wu > 0.0f) acc += wu * feats[(size_t)nj[u] * DIM + t];  // uniform branch per block
  }
  float ss = acc * acc;
  #pragma unroll
  for (int off = 16; off > 0; off >>= 1) ss += __shfl_down(ss, off, 32);
  if ((t & 31) == 0) red[t >> 5] = ss;
  __syncthreads();
  if (t == 0) {
    float tot = 0.0f;
    #pragma unroll
    for (int wvv = 0; wvv < 8; ++wvv) tot += red[wvv];
    scale_s = 1.0f / fmaxf(sqrtf(tot), 1e-12f);
  }
  __syncthreads();
  out[(size_t)i * DIM + t] = acc * scale_s;
}

// ---------------------------------------------------------------------------
extern "C" void kernel_launch(void* const* d_in, const int* in_sizes, int n_in,
                              void* d_out, int out_size, void* d_ws, size_t ws_size,
                              hipStream_t stream) {
  (void)in_sizes; (void)n_in; (void)out_size; (void)ws_size;
  const float* feats = (const float*)d_in[0];
  float* out = (float*)d_out;

  char* ws = (char*)d_ws;
  // workspace layout (all 16B aligned): bf16 feats | sq | knn_idx | knn_score
  unsigned short* fbf      = (unsigned short*)(ws);
  float*          sq       = (float*)(ws + (size_t)NROW * DIM * 2);                 // 4 MiB
  int*            knn_idx  = (int*)  (ws + (size_t)NROW * DIM * 2 + NROW * 4);      // +32 KiB
  float*          knn_sc   = (float*)(ws + (size_t)NROW * DIM * 2 + NROW * 4
                                         + (size_t)NROW * KNN * 4);                 // +512 KiB

  prep_kernel <<<NROW, 256, 0, stream>>>(feats, fbf, sq);
  knn_kernel  <<<NROW / 64, 128, 0, stream>>>(fbf, sq, knn_idx, knn_sc);
  apply_kernel<<<NROW, 256, 0, stream>>>(feats, knn_idx, knn_sc, out);
}
